// sLSTMCell_72911364817341
// MI455X (gfx1250) — matrix operand order repaired and checked by reading, not compiled, not run on hardware
//
#include <hip/hip_runtime.h>
#include <math.h>

// ---------------- problem constants ----------------
#define NHEAD 4
#define HD    512
#define EDIM  2048        // NHEAD*HD
#define GG    4
#define BSZ   32
#define SEQ   512
#define GE    8192        // GG*EDIM  (x last dim)

// ---------------- decomposition ----------------
#define WPH   16          // workgroups per head
#define DCH   32          // hidden dims owned per workgroup
#define NCOL  128         // GG*DCH columns of R per workgroup
#define KST   16          // K steps: 512 / 32
#define HSTR  520         // padded LDS row stride for h tile (1040 B = 65*16, keeps B128 alignment)
#define NBLK  (NHEAD*WPH) // 64 workgroups total
#define NTHR  256         // 8 waves of 32

// ---------------- LDS layout (bytes) ----------------
#define RF_BYTES   (512*NCOL*2)       // 131072 : R slice, WMMA-fragment order
#define HL_BYTES   (BSZ*HSTR*2)       // 33280  : h tile (A matrix), bf16, padded
#define RAW_BYTES  (BSZ*NCOL*4)       // 16384  : raw gate pre-activations, f32
#define BIAS_BYTES (NCOL*4)           // 512
#define SMEM_BYTES (RF_BYTES + HL_BYTES + RAW_BYTES + BIAS_BYTES)

typedef __bf16 v16bf __attribute__((ext_vector_type(16)));
typedef __bf16 v8bf  __attribute__((ext_vector_type(8)));
typedef float  v8f   __attribute__((ext_vector_type(8)));

// bf16 round-to-nearest-even helpers (bit-exact, no reliance on __bf16 casts)
__device__ __forceinline__ __bf16 f2b(float f) {
  unsigned u = __float_as_uint(f);
  unsigned r = (u + 0x7fffu + ((u >> 16) & 1u)) >> 16;
  unsigned short s = (unsigned short)r;
  __bf16 b; __builtin_memcpy(&b, &s, 2); return b;
}
__device__ __forceinline__ float b2f(__bf16 b) {
  unsigned short s; __builtin_memcpy(&s, &b, 2);
  return __uint_as_float(((unsigned)s) << 16);
}
__device__ __forceinline__ float rb(float f) { return b2f(f2b(f)); }

// Async global -> LDS 128-bit copy (CDNA5 GLOBAL_LOAD_ASYNC_TO_LDS_B128, ASYNCcnt).
// ldsoff = wave-relative LDS byte offset (low 32 bits of generic shared address).
__device__ __forceinline__ void async_g2l_b128(unsigned ldsoff, const void* gaddr) {
  asm volatile("global_load_async_to_lds_b128 %0, %1, off"
               :: "v"(ldsoff), "v"((unsigned long long)(uintptr_t)gaddr)
               : "memory");
}
__device__ __forceinline__ void wait_asynccnt0() {
  asm volatile("s_wait_asynccnt 0x0" ::: "memory");
}

// A-matrix fragment loader: 16-bit A 16x32 layout (ISA 7.12.2).
// lane<16 holds K = k0+{0..7,16..23}; lane>=16 holds K = k0+{8..15,24..31}.
__device__ __forceinline__ v16bf load_a(const __bf16* hl, int row, int k0, int half) {
  const __bf16* p = hl + row * HSTR + k0 + half * 8;
  v8bf lo = *(const v8bf*)p;         // VGPR0..3  : K pairs
  v8bf hi = *(const v8bf*)(p + 16);  // VGPR4..7  : K pairs +16
  return __builtin_shufflevector(lo, hi, 0,1,2,3,4,5,6,7,8,9,10,11,12,13,14,15);
}

// Per-head barrier: only the 16 workgroups of one head synchronize.
__device__ __forceinline__ void head_barrier(unsigned int* cnt, unsigned int target) {
  __syncthreads();
  if (threadIdx.x == 0) {
    __threadfence();
    __hip_atomic_fetch_add(cnt, 1u, __ATOMIC_ACQ_REL, __HIP_MEMORY_SCOPE_AGENT);
    while (__hip_atomic_load(cnt, __ATOMIC_ACQUIRE, __HIP_MEMORY_SCOPE_AGENT) < target) {
      __builtin_amdgcn_s_sleep(8);
    }
    __threadfence();
  }
  __syncthreads();
}

__global__ void __launch_bounds__(NTHR, 1)
slstm_persistent(const float* __restrict__ x,      // (32,512,8192)
                 const float* __restrict__ st,     // (4,32,2048)  y,c,n,m
                 const float* __restrict__ R,      // (4,512,2048)
                 const float* __restrict__ bias,   // (8192) = (NH,G,HD)
                 float* __restrict__ out,          // out(32,4,512,512) ++ final(4,32,2048)
                 __bf16* __restrict__ hbuf,        // double-buffered (2,4,32,512) bf16
                 unsigned int* __restrict__ cnts)  // 4 per-head counters, 64B apart
{
  extern __shared__ char smem[];
  __bf16* Rf    = (__bf16*)(smem);
  __bf16* hl    = (__bf16*)(smem + RF_BYTES);
  float*  rawl  = (float*) (smem + RF_BYTES + HL_BYTES);
  float*  biasl = (float*) (smem + RF_BYTES + HL_BYTES + RAW_BYTES);

  const int tid  = threadIdx.x;
  const int w    = tid >> 5;          // wave id == N-tile id (0..7)
  const int lane = tid & 31;
  const int mlane = lane & 15;
  const int half  = lane >> 4;

  const int h  = blockIdx.x >> 4;     // head
  const int j  = blockIdx.x & 15;     // dim-chunk within head
  const int d0 = j * DCH;
  unsigned int* cnt = cnts + h * 16;  // 64-byte separation per head

  // ---- one-time: bias slice into LDS ----
  if (tid < NCOL) {
    int g = tid >> 5, i = tid & 31;
    biasl[tid] = bias[h * (GG * HD) + g * HD + d0 + i];
  }

  // ---- one-time: pre-swizzle R slice into WMMA B-fragment layout ----
  // fragment f = ntile*16 + ks ; per lane 16 bf16 at Rf[(f*32+lane)*16 + e]
  // B layout: col = ntile*16 + lane%16 ; K = ks*32 + 8*(lane/16) + (e%8) + 16*(e/8)
  for (int f = w; f < 8 * KST; f += 8) {
    int ntile = f >> 4, ks = f & 15;
    int ncol  = ntile * 16 + mlane;
    int g = ncol >> 5, i = ncol & 31;
    size_t col = (size_t)(g * HD + d0 + i);
    #pragma unroll
    for (int e = 0; e < 16; e++) {
      int k = ks * 32 + half * 8 + (e & 7) + ((e >> 3) << 4);
      Rf[(((size_t)f * 32 + lane) << 4) + e] =
          f2b(R[((size_t)(h * HD + k)) * (size_t)(GG * HD) + col]);
    }
  }

  // ---- per-thread persistent carry: 4 (b,d) items each ----
  float cs[4], ns[4], ms[4], hs[4];
  #pragma unroll
  for (int r = 0; r < 4; r++) {
    int b = 8 * r + (tid >> 5);
    int d = d0 + (tid & 31);
    size_t o = (size_t)b * EDIM + h * HD + d;
    cs[r] = rb(st[(size_t)1 * BSZ * EDIM + o]);
    ns[r] = rb(st[(size_t)2 * BSZ * EDIM + o]);
    ms[r] = rb(st[(size_t)3 * BSZ * EDIM + o]);
    hs[r] = rb(st[o]);
  }
  __syncthreads();

  const size_t HBUF = (size_t)NHEAD * BSZ * HD;
  const unsigned hl_ldsoff = (unsigned)(uintptr_t)(void*)hl;   // wave-relative LDS byte offset

  for (int t = 0; t < SEQ; t++) {
    const __bf16* hread  = hbuf + (size_t)(t & 1) * HBUF;
    __bf16*       hwrite = hbuf + (size_t)((t + 1) & 1) * HBUF;

    // ---- phase 1: stage full head h-state into LDS (A matrix) ----
    if (t == 0) {
      // initial state: f32 -> bf16 conversion, manual copy (runs once)
      for (int e = tid; e < BSZ * HD; e += NTHR) {
        int b = e >> 9, d = e & 511;
        hl[b * HSTR + d] = f2b(st[(size_t)b * EDIM + h * HD + d]);
      }
    } else {
      // steady state: async global->LDS B128 copies (8 bf16 per transfer)
      #pragma unroll
      for (int v = 0; v < (BSZ * HD / 8) / NTHR; v++) {     // 8 iterations
        int vi = tid + NTHR * v;                            // 128-bit chunk index
        int b = vi >> 6;                                    // 64 chunks per batch row
        int d = (vi & 63) * 8;
        unsigned loff = hl_ldsoff + (unsigned)(b * HSTR + d) * 2u;
        const void* g = (const void*)(hread + (size_t)(h * BSZ + b) * HD + d);
        async_g2l_b128(loff, g);
      }
      wait_asynccnt0();
    }
    __syncthreads();

    // ---- phase 2: GEMM  (32x512)@(512x128) via v_wmma_f32_16x16x32_bf16 ----
    v8f acc0 = {}; v8f acc1 = {};
    #pragma unroll
    for (int ks = 0; ks < KST; ks++) {
      int k0 = ks * 32;
      v16bf a0 = load_a(hl, mlane,      k0, half);
      v16bf a1 = load_a(hl, 16 + mlane, k0, half);
      v16bf bm = *(const v16bf*)&Rf[(((size_t)(w * KST + ks) * 32 + lane)) << 4];
      acc0 = __builtin_amdgcn_wmma_f32_16x16x32_bf16(false, a0, false, bm, (short)0, acc0, false, false);
      acc1 = __builtin_amdgcn_wmma_f32_16x16x32_bf16(false, a1, false, bm, (short)0, acc1, false, false);
    }

    // ---- phase 3: spill accumulators (rounded to bf16 like jnp einsum) ----
    #pragma unroll
    for (int v = 0; v < 8; v++) {
      rawl[(8 * half + v)        * NCOL + w * 16 + mlane] = rb(acc0[v]);
      rawl[(16 + 8 * half + v)   * NCOL + w * 16 + mlane] = rb(acc1[v]);
    }
    __syncthreads();

    // ---- phase 4: gate math, carry update, outputs ----
    #pragma unroll
    for (int r = 0; r < 4; r++) {
      int b = 8 * r + (tid >> 5);
      int i = tid & 31;
      int d = d0 + i;
      const float* xp = x + ((size_t)b * SEQ + t) * GE + h * HD + d;
      if (t + 1 < SEQ) __builtin_prefetch(xp + GE, 0, 1);

      float ir = rawl[b * NCOL + 0 * DCH + i] + biasl[0 * DCH + i] + rb(xp[0 * EDIM]);
      float fr = rawl[b * NCOL + 1 * DCH + i] + biasl[1 * DCH + i] + rb(xp[1 * EDIM]);
      float zr = rawl[b * NCOL + 2 * DCH + i] + biasl[2 * DCH + i] + rb(xp[2 * EDIM]);
      float orv= rawl[b * NCOL + 3 * DCH + i] + biasl[3 * DCH + i] + rb(xp[3 * EDIM]);

      float fc   = fminf(fr, 15.0f);
      float lsig = fminf(fc, 0.0f) - log1pf(expf(-fabsf(fc)));   // log_sigmoid
      float lfm  = ms[r] + lsig;
      float mnew = (t == 0) ? ir : fmaxf(ir, lfm);               // all(n==0) only at t=0

      float og = 1.0f / (1.0f + expf(-orv));
      float ig = fminf(expf(ir  - mnew), 1.0f);
      float fg = fminf(expf(lfm - mnew), 1.0f);
      float zg = tanhf(zr);

      float cn = fg * cs[r] + ig * zg;
      float nn = fg * ns[r] + ig;
      float hn = og * cn / nn;

      cs[r] = rb(cn); ns[r] = rb(nn); ms[r] = rb(mnew);
      __bf16 hb = f2b(hn);
      hs[r] = b2f(hb);

      hwrite[(size_t)(h * BSZ + b) * HD + d] = hb;
      out[((size_t)(b * NHEAD + h) * SEQ + t) * HD + d] = b2f(hb);
    }

    // ---- phase 5: per-head step barrier (16 workgroups) ----
    head_barrier(cnt, (unsigned)(t + 1) * WPH);
  }

  // ---- final state: (y,c,n,m) as f32 after output tensor ----
  const size_t FBASE = (size_t)BSZ * NHEAD * SEQ * HD;   // 33554432
  #pragma unroll
  for (int r = 0; r < 4; r++) {
    int b = 8 * r + (tid >> 5);
    int d = d0 + (tid & 31);
    size_t o = (size_t)b * EDIM + h * HD + d;
    out[FBASE + 0 * (size_t)BSZ * EDIM + o] = hs[r];
    out[FBASE + 1 * (size_t)BSZ * EDIM + o] = cs[r];
    out[FBASE + 2 * (size_t)BSZ * EDIM + o] = ns[r];
    out[FBASE + 3 * (size_t)BSZ * EDIM + o] = ms[r];
  }
}

extern "C" void kernel_launch(void* const* d_in, const int* in_sizes, int n_in,
                              void* d_out, int out_size, void* d_ws, size_t ws_size,
                              hipStream_t stream) {
  const float* x  = (const float*)d_in[0];
  const float* st = (const float*)d_in[1];
  const float* R  = (const float*)d_in[2];
  const float* b  = (const float*)d_in[3];
  float* out = (float*)d_out;

  unsigned int* cnts = (unsigned int*)d_ws;                // 4 counters, 64B apart
  __bf16* hbuf = (__bf16*)((char*)d_ws + 256);             // 2*4*32*512 bf16 = 256 KiB region

  // reset barrier counters every call (deterministic, graph-capture safe)
  hipMemsetAsync(d_ws, 0, 256, stream);

  hipFuncSetAttribute((const void*)slstm_persistent,
                      hipFuncAttributeMaxDynamicSharedMemorySize, SMEM_BYTES);

  slstm_persistent<<<NBLK, NTHR, SMEM_BYTES, stream>>>(x, st, R, b, out, hbuf, cnts);
}
